// NodeEncoder_25563645346495
// MI455X (gfx1250) — matrix operand order repaired and checked by reading, not compiled, not run on hardware
//
#include <hip/hip_runtime.h>
#include <hip/hip_bf16.h>

typedef _Float16 half_t;
typedef __attribute__((ext_vector_type(16))) _Float16 v16h;
typedef __attribute__((ext_vector_type(8)))  float    v8f;

#define HDIM 128

__device__ __forceinline__ float leaky02(float v) {
    return fmaxf(v, 0.0f) + 0.2f * fminf(v, 0.0f);
}

// ---------------------------------------------------------------------------
// Convert W1/W2 (L,H,H) f32 -> transposed f16: wt[l][n][k] = W[l][k][n]
// ---------------------------------------------------------------------------
__global__ void gin_wconv(const float* __restrict__ W1, const float* __restrict__ W2,
                          half_t* __restrict__ w1t, half_t* __restrict__ w2t, int total) {
    int idx = blockIdx.x * blockDim.x + threadIdx.x;
    if (idx >= total) return;                 // total = L*H*H
    int l = idx / (HDIM * HDIM);
    int r = idx - l * (HDIM * HDIM);
    int k = r / HDIM;
    int n = r - k * HDIM;
    int t = l * (HDIM * HDIM) + n * HDIM + k;
    w1t[t] = (half_t)W1[idx];
    w2t[t] = (half_t)W2[idx];
}

// ---------------------------------------------------------------------------
// h = x (float4 copy)
// ---------------------------------------------------------------------------
__global__ void gin_copy4(const float4* __restrict__ src, float4* __restrict__ dst, int n4) {
    int i = blockIdx.x * blockDim.x + threadIdx.x;
    if (i < n4) dst[i] = src[i];
}

// ---------------------------------------------------------------------------
// h[dst] += x[src] over edges. One wave per edge, 4 floats per lane.
// ---------------------------------------------------------------------------
__global__ void gin_scatter(const float4* __restrict__ x4, const int* __restrict__ esrc,
                            const int* __restrict__ edst, float* __restrict__ h, int n_edges) {
    int e    = (blockIdx.x * blockDim.x + threadIdx.x) >> 5;
    int lane = threadIdx.x & 31;
    if (e >= n_edges) return;
    int s = esrc[e];
    int d = edst[e];
    float4 v = x4[(size_t)s * (HDIM / 4) + lane];
    float* hp = h + (size_t)d * HDIM + lane * 4;
    atomicAdd(hp + 0, v.x);
    atomicAdd(hp + 1, v.y);
    atomicAdd(hp + 2, v.z);
    atomicAdd(hp + 3, v.w);
}

// ---------------------------------------------------------------------------
// Fused GIN MLP: xout = leaky(leaky(h @ W1 + b1) @ W2 + b2)
// 256 threads = 8 waves, each wave owns a 16-row tile, WMMA f32_16x16x32_f16.
// ---------------------------------------------------------------------------
__global__ __launch_bounds__(256)
void gin_mlp(const float* __restrict__ h,
             const half_t* __restrict__ w1t, const float* __restrict__ b1,
             const half_t* __restrict__ w2t, const float* __restrict__ b2,
             float* __restrict__ xout, int n_nodes) {
    __shared__ half_t tstage[8][16][HDIM + 8];      // padded rows (272B stride)

    const int wave = threadIdx.x >> 5;
    const int lane = threadIdx.x & 31;
    const int ll   = lane & 15;                     // A: row / B,C,D: column
    const int hi   = lane >> 4;                     // K-half / M-half selector
    const int row0 = blockIdx.x * 128 + wave * 16;

    // ---- A fragments for GEMM1: rows of h, f32 -> f16 (clamped, EXEC stays full)
    int arow = row0 + ll;
    if (arow >= n_nodes) arow = n_nodes - 1;
    const float* hrow = h + (size_t)arow * HDIM;
    v16h a[4];
#pragma unroll
    for (int k0 = 0; k0 < 4; ++k0) {
        const float* p0 = hrow + k0 * 32 + hi * 16;  // K = k0*32 + hi*16 + j
        const float* p1 = p0 + 8;                    // K = k0*32 + hi*16 + 8 + j
#pragma unroll
        for (int j = 0; j < 8; ++j) {
            a[k0][j]     = (half_t)p0[j];
            a[k0][8 + j] = (half_t)p1[j];
        }
    }

    half_t* tw = &tstage[wave][0][0];

    // ---- GEMM1 + bias + leaky, stage 16x128 f16 tile in LDS
#pragma unroll
    for (int n0 = 0; n0 < 8; ++n0) {
        v8f c = {};
#pragma unroll
        for (int k0 = 0; k0 < 4; ++k0) {
            const v16h* bp = (const v16h*)(w1t + (size_t)(n0 * 16 + ll) * HDIM + k0 * 32 + hi * 16);
            c = __builtin_amdgcn_wmma_f32_16x16x32_f16(false, a[k0], false, *bp,
                                                       (short)0, c, false, false);
        }
        float bias = b1[n0 * 16 + ll];
#pragma unroll
        for (int r = 0; r < 8; ++r) {
            float v = leaky02(c[r] + bias);
            tw[(r + hi * 8) * (HDIM + 8) + (n0 * 16 + ll)] = (half_t)v;
        }
    }

    // per-wave LDS RAW fence: wait for all DS stores before reloading
    asm volatile("s_wait_dscnt 0" ::: "memory");

    // ---- A fragments for GEMM2 from staged tile
    v16h a2[4];
#pragma unroll
    for (int k0 = 0; k0 < 4; ++k0) {
        const half_t* q0 = &tw[ll * (HDIM + 8) + k0 * 32 + hi * 16];
        const half_t* q1 = q0 + 8;
#pragma unroll
        for (int j = 0; j < 8; ++j) {
            a2[k0][j]     = q0[j];
            a2[k0][8 + j] = q1[j];
        }
    }

    // ---- GEMM2 + bias + leaky, store f32
#pragma unroll
    for (int n0 = 0; n0 < 8; ++n0) {
        v8f c = {};
#pragma unroll
        for (int k0 = 0; k0 < 4; ++k0) {
            const v16h* bp = (const v16h*)(w2t + (size_t)(n0 * 16 + ll) * HDIM + k0 * 32 + hi * 16);
            c = __builtin_amdgcn_wmma_f32_16x16x32_f16(false, a2[k0], false, *bp,
                                                       (short)0, c, false, false);
        }
        float bias = b2[n0 * 16 + ll];
#pragma unroll
        for (int r = 0; r < 8; ++r) {
            float v = leaky02(c[r] + bias);
            int row = row0 + r + hi * 8;
            if (row < n_nodes) xout[(size_t)row * HDIM + n0 * 16 + ll] = v;
        }
    }
}

// ---------------------------------------------------------------------------
// LayerNorm over feature dim: one wave per node (wave32, 4 feats/lane)
// ---------------------------------------------------------------------------
__global__ void gin_layernorm(const float* __restrict__ xin, const float* __restrict__ gam,
                              const float* __restrict__ bet, float* __restrict__ out, int n_nodes) {
    int node = (blockIdx.x * blockDim.x + threadIdx.x) >> 5;
    int lane = threadIdx.x & 31;
    if (node >= n_nodes) return;
    const float4* p = (const float4*)(xin + (size_t)node * HDIM);
    float4 v = p[lane];
    float s  = v.x + v.y + v.z + v.w;
    float ss = v.x * v.x + v.y * v.y + v.z * v.z + v.w * v.w;
#pragma unroll
    for (int off = 16; off > 0; off >>= 1) {
        s  += __shfl_xor(s, off, 32);
        ss += __shfl_xor(ss, off, 32);
    }
    float mu  = s * (1.0f / HDIM);
    float var = ss * (1.0f / HDIM) - mu * mu;
    float inv = rsqrtf(var + 1e-5f);
    float4 g = ((const float4*)gam)[lane];
    float4 b = ((const float4*)bet)[lane];
    float4 o;
    o.x = (v.x - mu) * inv * g.x + b.x;
    o.y = (v.y - mu) * inv * g.y + b.y;
    o.z = (v.z - mu) * inv * g.z + b.z;
    o.w = (v.w - mu) * inv * g.w + b.w;
    ((float4*)(out + (size_t)node * HDIM))[lane] = o;
}

__global__ void gin_zero(float* __restrict__ p, int n) {
    int i = blockIdx.x * blockDim.x + threadIdx.x;
    if (i < n) p[i] = 0.0f;
}

// ---------------------------------------------------------------------------
// global_add_pool: one wave per node, atomicAdd into g[batch[node]]
// ---------------------------------------------------------------------------
__global__ void gin_pool(const float* __restrict__ emb, const int* __restrict__ batch,
                         float* __restrict__ g, int n_nodes) {
    int node = (blockIdx.x * blockDim.x + threadIdx.x) >> 5;
    int lane = threadIdx.x & 31;
    if (node >= n_nodes) return;
    int b = batch[node];
    float4 v = ((const float4*)(emb + (size_t)node * HDIM))[lane];
    float* gp = g + (size_t)b * HDIM + lane * 4;
    atomicAdd(gp + 0, v.x);
    atomicAdd(gp + 1, v.y);
    atomicAdd(gp + 2, v.z);
    atomicAdd(gp + 3, v.w);
}

// ---------------------------------------------------------------------------
// BatchNorm over the graph batch dim: 1 block, thread = feature
// ---------------------------------------------------------------------------
__global__ void gin_bn(const float* __restrict__ g, const float* __restrict__ gam,
                       const float* __restrict__ bet, float* __restrict__ out, int G) {
    int f = threadIdx.x;
    float s = 0.0f, ss = 0.0f;
    for (int i = 0; i < G; ++i) {
        float v = g[(size_t)i * HDIM + f];
        s += v; ss += v * v;
    }
    float mu  = s / (float)G;
    float var = ss / (float)G - mu * mu;
    float inv = rsqrtf(var + 1e-5f);
    float ga = gam[f], be = bet[f];
    for (int i = 0; i < G; ++i)
        out[(size_t)i * HDIM + f] = (g[(size_t)i * HDIM + f] - mu) * inv * ga + be;
}

// ---------------------------------------------------------------------------
extern "C" void kernel_launch(void* const* d_in, const int* in_sizes, int n_in,
                              void* d_out, int out_size, void* d_ws, size_t ws_size,
                              hipStream_t stream) {
    const float* x    = (const float*)d_in[0];
    const int*   ei   = (const int*)d_in[1];
    const int*   bat  = (const int*)d_in[2];
    const float* W1   = (const float*)d_in[3];
    const float* b1   = (const float*)d_in[4];
    const float* W2   = (const float*)d_in[5];
    const float* b2   = (const float*)d_in[6];
    const float* ln_g = (const float*)d_in[7];
    const float* ln_b = (const float*)d_in[8];
    const float* bn_g = (const float*)d_in[9];
    const float* bn_b = (const float*)d_in[10];

    const int n_nodes = in_sizes[0] / HDIM;
    const int n_edges = in_sizes[1] / 2;
    const int L       = in_sizes[3] / (HDIM * HDIM);
    const int G       = (out_size - n_nodes * HDIM) / HDIM;
    const int* esrc = ei;
    const int* edst = ei + n_edges;

    // workspace carve-up (256B aligned)
    size_t nh_bytes = (size_t)n_nodes * HDIM * sizeof(float);
    nh_bytes = (nh_bytes + 255) & ~(size_t)255;
    char* ws = (char*)d_ws;
    float*  bufA = (float*)ws;                 ws += nh_bytes;       // MLP output / cur x
    float*  bufB = (float*)ws;                 ws += nh_bytes;       // h = x + agg
    float*  gbuf = (float*)ws;                 ws += ((size_t)G * HDIM * sizeof(float) + 255) & ~(size_t)255;
    half_t* w1t  = (half_t*)ws;                ws += ((size_t)L * HDIM * HDIM * sizeof(half_t) + 255) & ~(size_t)255;
    half_t* w2t  = (half_t*)ws;

    const int wtot = L * HDIM * HDIM;
    gin_wconv<<<(wtot + 255) / 256, 256, 0, stream>>>(W1, W2, w1t, w2t, wtot);

    const int n4        = n_nodes * (HDIM / 4);
    const int copy_grid = (n4 + 255) / 256;
    const int scat_grid = (int)(((long long)n_edges * 32 + 255) / 256);
    const int mlp_grid  = (n_nodes + 127) / 128;
    const int node_grid = (int)(((long long)n_nodes * 32 + 255) / 256);

    const float* cur = x;
    for (int l = 0; l < L; ++l) {
        gin_copy4<<<copy_grid, 256, 0, stream>>>((const float4*)cur, (float4*)bufB, n4);
        gin_scatter<<<scat_grid, 256, 0, stream>>>((const float4*)cur, esrc, edst, bufB, n_edges);
        gin_mlp<<<mlp_grid, 256, 0, stream>>>(bufB,
                                              w1t + (size_t)l * HDIM * HDIM, b1 + (size_t)l * HDIM,
                                              w2t + (size_t)l * HDIM * HDIM, b2 + (size_t)l * HDIM,
                                              bufA, n_nodes);
        cur = bufA;
    }

    float* node_emb  = (float*)d_out;
    float* graph_emb = (float*)d_out + (size_t)n_nodes * HDIM;

    gin_layernorm<<<node_grid, 256, 0, stream>>>(cur, ln_g, ln_b, node_emb, n_nodes);
    gin_zero<<<(G * HDIM + 255) / 256, 256, 0, stream>>>(gbuf, G * HDIM);
    gin_pool<<<node_grid, 256, 0, stream>>>(node_emb, bat, gbuf, n_nodes);
    gin_bn<<<1, HDIM, 0, stream>>>(gbuf, bn_g, bn_b, graph_emb, G);
}
